// RelativePositionalMultiHeadAttention_87711822119152
// MI455X (gfx1250) — matrix operand order repaired and checked by reading, not compiled, not run on hardware
//
#include <hip/hip_runtime.h>
#include <hip/hip_bf16.h>

#define IN_CH   512
#define HEADS   8
#define HEAD_CH 64
#define WIN     4
#define BATCH   8
#define SEQ     1024

typedef __bf16 bf16;
typedef __attribute__((ext_vector_type(16))) __bf16 v16bf;
typedef __attribute__((ext_vector_type(8)))  float  v8f;
typedef unsigned int u32x4 __attribute__((ext_vector_type(4)));
typedef int          i32x8 __attribute__((ext_vector_type(8)));
typedef int          i32x4 __attribute__((ext_vector_type(4)));

static __device__ inline v8f zero8() {
    v8f z;
#pragma unroll
    for (int i = 0; i < 8; ++i) z[i] = 0.0f;
    return z;
}

// A-fragment (16x32 bf16, M x K): lane ln=lane&15 holds row M=ln, hi=lane>>4.
// elements j <- K = hi*16 + j (16 contiguous bf16 from base = p0 - hi*8)
static __device__ inline v16bf ld_frag_a(const bf16* p0, const bf16* p1) {
    v16bf f;
    int hh = (threadIdx.x & 31) >> 4;
    const bf16* base = p0 - hh * 8;
#pragma unroll
    for (int j = 0; j < 8; ++j) { f[j] = base[hh * 16 + j]; f[j + 8] = base[hh * 16 + 8 + j]; }
    return f;
}

// B-fragment (32x16 bf16, K x N) loaded from B^T (row-major [N][K]):
// lane ln holds column N=ln; elements j <- K = hi*16 + j (16 contiguous bf16).
static __device__ inline v16bf ld_frag_b(const bf16* p) {
    v16bf f;
#pragma unroll
    for (int j = 0; j < 16; ++j) f[j] = p[j];
    return f;
}

static __device__ inline v8f wmma_bf16(v16bf a, v16bf b, v8f c) {
    return __builtin_amdgcn_wmma_f32_16x16x32_bf16(
        false, a, false, b, (short)0, c, false, false);
}

// TDM: async DMA of a 2-D tile (tile_s contiguous elems x tile_d rows, row
// stride stride_elems, 2-byte elements) from global into LDS at lds_off.
// Descriptor packing per CDNA5 ISA (groups 0/1).
// This toolchain exposes the 6-arg builtin (g0, g1, g2, g3, g4, cpol).
static __device__ inline void tdm_load_2d(unsigned lds_off, const bf16* gptr,
                                          int tile_s, int tile_d, int stride_elems) {
    unsigned long long ga = (unsigned long long)(size_t)gptr;
    u32x4 g0;
    g0[0] = 1u;                                            // count=1, user mode
    g0[1] = lds_off;                                       // lds_addr (bytes)
    g0[2] = (unsigned)(ga & 0xffffffffu);                  // global_addr[31:0]
    g0[3] = (unsigned)((ga >> 32) & 0x1ffffffu) | (2u << 30); // addr[56:32], type=2
    i32x8 g1;
    g1[0] = 1 << 16;                                       // data_size=1 (2 bytes)
    g1[1] = (tile_s & 0xffff) << 16;                       // tensor_dim0 lo16
    g1[2] = ((tile_s >> 16) & 0xffff) | ((tile_d & 0xffff) << 16); // d0 hi, d1 lo
    g1[3] = ((tile_d >> 16) & 0xffff) | ((tile_s & 0xffff) << 16); // d1 hi, tile_dim0
    g1[4] = (tile_d & 0xffff);                             // tile_dim1 (tile_dim2=0)
    g1[5] = stride_elems;                                  // tensor_dim0_stride lo32
    g1[6] = 0;
    g1[7] = 0;
    i32x4 z4;
#pragma unroll
    for (int i = 0; i < 4; ++i) z4[i] = 0;
    i32x8 z8;
#pragma unroll
    for (int i = 0; i < 8; ++i) z8[i] = 0;
    __builtin_amdgcn_tensor_load_to_lds(g0, g1, z4, z4, z8, 0);
}

// ---------------------------------------------------------------- converts
__global__ void cvt_f32_bf16(const float* __restrict__ in, bf16* __restrict__ out, int n) {
    int i = blockIdx.x * blockDim.x + threadIdx.x;
    if (i < n) out[i] = (bf16)in[i];
}

// [b][c][s] f32 -> [b][s][c] bf16  (LDS 32x32 tile transpose)
__global__ void transpose_cvt(const float* __restrict__ in, bf16* __restrict__ out) {
    __shared__ float tile[32][33];
    int b = blockIdx.z;
    int c0 = blockIdx.y * 32, s0 = blockIdx.x * 32;
    int tx = threadIdx.x, ty = threadIdx.y;          // (32, 8)
    const float* src = in + (size_t)b * IN_CH * SEQ;
#pragma unroll
    for (int i = 0; i < 32; i += 8)
        tile[ty + i][tx] = src[(size_t)(c0 + ty + i) * SEQ + s0 + tx];
    __syncthreads();
    bf16* dst = out + (size_t)b * SEQ * IN_CH;
#pragma unroll
    for (int i = 0; i < 32; i += 8)
        dst[(size_t)(s0 + ty + i) * IN_CH + c0 + tx] = (bf16)tile[tx][ty + i];
}

// ---------------------------------------------------------------- projections
// Out = W(512x512) @ X_b(512x1024) + bias.  XT is [b][s][i] bf16 (acts as B^T).
// Wave tile: 32(M) x 64(N); K-loop ping-pong software pipelined (no reg copies).
// vlayout==0: store [b][h][s][d] (Q,K);  vlayout==1: store [b][o][s] (V)
__global__ void __launch_bounds__(128, 1)
proj_gemm(const bf16* __restrict__ W, const bf16* __restrict__ XT,
          const float* __restrict__ bias, bf16* __restrict__ out, int vlayout) {
    int lane = threadIdx.x & 31;
    int wid  = threadIdx.x >> 5;
    int ln = lane & 15, hi = lane >> 4;
    int b  = blockIdx.z;
    int o0 = blockIdx.y * 128 + wid * 32;
    int s0 = blockIdx.x * 64;
    const bf16* w0 = W + (size_t)(o0 + ln) * IN_CH;
    const bf16* w1 = W + (size_t)(o0 + 16 + ln) * IN_CH;
    const bf16* xb = XT + (size_t)b * SEQ * IN_CH + (size_t)ln * IN_CH;

    v8f acc[2][4];
#pragma unroll
    for (int mt = 0; mt < 2; ++mt)
#pragma unroll
        for (int nt = 0; nt < 4; ++nt) acc[mt][nt] = zero8();

    auto loadA = [&](v16bf* a, int k0) {
        a[0] = ld_frag_a(w0 + k0 + hi * 8, w0 + k0 + 16 + hi * 8);
        a[1] = ld_frag_a(w1 + k0 + hi * 8, w1 + k0 + 16 + hi * 8);
    };
    auto loadB = [&](v16bf* bb, int k0) {
#pragma unroll
        for (int nt = 0; nt < 4; ++nt)
            bb[nt] = ld_frag_b(xb + (size_t)(s0 + nt * 16) * IN_CH + k0 + hi * 16);
    };

    v16bf aA[2], bA[4], aB[2], bB[4];
    loadA(aA, 0); loadB(bA, 0);
#pragma unroll 1
    for (int k0 = 0; k0 < IN_CH; k0 += 64) {
        loadA(aB, k0 + 32); loadB(bB, k0 + 32);
#pragma unroll
        for (int mt = 0; mt < 2; ++mt)
#pragma unroll
            for (int nt = 0; nt < 4; ++nt)
                acc[mt][nt] = wmma_bf16(aA[mt], bA[nt], acc[mt][nt]);
        if (k0 + 64 < IN_CH) { loadA(aA, k0 + 64); loadB(bA, k0 + 64); }
#pragma unroll
        for (int mt = 0; mt < 2; ++mt)
#pragma unroll
            for (int nt = 0; nt < 4; ++nt)
                acc[mt][nt] = wmma_bf16(aB[mt], bB[nt], acc[mt][nt]);
    }

#pragma unroll
    for (int mt = 0; mt < 2; ++mt)
#pragma unroll
        for (int nt = 0; nt < 4; ++nt)
#pragma unroll
            for (int v = 0; v < 8; ++v) {
                int o = o0 + mt * 16 + v + 8 * hi;
                int s = s0 + nt * 16 + ln;
                float val = acc[mt][nt][v] + bias[o];
                size_t addr;
                if (vlayout) {
                    addr = ((size_t)b * IN_CH + o) * SEQ + s;                   // [b][o][s]
                } else {
                    int h = o >> 6, d = o & 63;                                 // o=h*64+d
                    addr = (((size_t)(b * HEADS + h)) * SEQ + s) * HEAD_CH + d; // [b][h][s][d]
                }
                out[addr] = (bf16)val;
            }
}

// out = Wo @ ctx + bo, ctx given as [b][s][i] bf16 (B^T), f32 store [b][o][s]
__global__ void __launch_bounds__(128, 1)
out_gemm(const bf16* __restrict__ W, const bf16* __restrict__ CT,
         const float* __restrict__ bias, float* __restrict__ out) {
    int lane = threadIdx.x & 31;
    int wid  = threadIdx.x >> 5;
    int ln = lane & 15, hi = lane >> 4;
    int b  = blockIdx.z;
    int o0 = blockIdx.y * 128 + wid * 32;
    int s0 = blockIdx.x * 64;
    const bf16* w0 = W + (size_t)(o0 + ln) * IN_CH;
    const bf16* w1 = W + (size_t)(o0 + 16 + ln) * IN_CH;
    const bf16* xb = CT + (size_t)b * SEQ * IN_CH + (size_t)ln * IN_CH;

    v8f acc[2][4];
#pragma unroll
    for (int mt = 0; mt < 2; ++mt)
#pragma unroll
        for (int nt = 0; nt < 4; ++nt) acc[mt][nt] = zero8();

    auto loadA = [&](v16bf* a, int k0) {
        a[0] = ld_frag_a(w0 + k0 + hi * 8, w0 + k0 + 16 + hi * 8);
        a[1] = ld_frag_a(w1 + k0 + hi * 8, w1 + k0 + 16 + hi * 8);
    };
    auto loadB = [&](v16bf* bb, int k0) {
#pragma unroll
        for (int nt = 0; nt < 4; ++nt)
            bb[nt] = ld_frag_b(xb + (size_t)(s0 + nt * 16) * IN_CH + k0 + hi * 16);
    };

    v16bf aA[2], bA[4], aB[2], bB[4];
    loadA(aA, 0); loadB(bA, 0);
#pragma unroll 1
    for (int k0 = 0; k0 < IN_CH; k0 += 64) {
        loadA(aB, k0 + 32); loadB(bB, k0 + 32);
#pragma unroll
        for (int mt = 0; mt < 2; ++mt)
#pragma unroll
            for (int nt = 0; nt < 4; ++nt)
                acc[mt][nt] = wmma_bf16(aA[mt], bA[nt], acc[mt][nt]);
        if (k0 + 64 < IN_CH) { loadA(aA, k0 + 64); loadB(bA, k0 + 64); }
#pragma unroll
        for (int mt = 0; mt < 2; ++mt)
#pragma unroll
            for (int nt = 0; nt < 4; ++nt)
                acc[mt][nt] = wmma_bf16(aB[mt], bB[nt], acc[mt][nt]);
    }

#pragma unroll
    for (int mt = 0; mt < 2; ++mt)
#pragma unroll
        for (int nt = 0; nt < 4; ++nt)
#pragma unroll
            for (int v = 0; v < 8; ++v) {
                int o = o0 + mt * 16 + v + 8 * hi;
                int s = s0 + nt * 16 + ln;
                out[((size_t)b * IN_CH + o) * SEQ + s] = acc[mt][nt][v] + bias[o];
            }
}

// ---------------------------------------------------------------- attention
// One wave per (b, h, 16-query tile). Flash-style online softmax over 32-key
// tiles, two tiles per loop iteration (ping-pong registers / LDS buffers).
// K fragments register-double-buffered; V tiles staged into LDS by the Tensor
// Data Mover (double-buffered, s_wait_tensorcnt-synchronized).
// Q,K: [b][h][s][d] bf16; V: [b][h][d][s] bf16; ctx out: [b][s][c] bf16.
__global__ void __launch_bounds__(32, 1)
attn_kernel(const bf16* __restrict__ Q, const bf16* __restrict__ K,
            const bf16* __restrict__ V, const float* __restrict__ mask,
            const float* __restrict__ relk, const float* __restrict__ relv,
            bf16* __restrict__ ctx) {
    __shared__ bf16  Vt[2][HEAD_CH][32];   // two 64x32 V tiles (TDM destination)
    __shared__ bf16  P_lds[16][32];
    __shared__ float relS[16][16];

    int lane = threadIdx.x & 31;
    int ln = lane & 15, hi = lane >> 4;
    int q0 = blockIdx.x * 16;
    int h  = blockIdx.y;
    int b  = blockIdx.z;

    const bf16* qh = Q + ((size_t)(b * HEADS + h)) * SEQ * HEAD_CH;
    const bf16* kh = K + ((size_t)(b * HEADS + h)) * SEQ * HEAD_CH;
    const bf16* vh = V + ((size_t)(b * HEADS + h)) * HEAD_CH * SEQ;   // [d][s]
    unsigned vt_off0 = (unsigned)(size_t)&Vt[0][0][0];
    unsigned vt_off1 = (unsigned)(size_t)&Vt[1][0][0];

    // Q A-fragments for the two 32-wide halves of d
    v16bf qf[2];
#pragma unroll
    for (int dc = 0; dc < 2; ++dc) {
        const bf16* r = qh + (size_t)(q0 + ln) * HEAD_CH + dc * 32;
        qf[dc] = ld_frag_a(r + hi * 8, r + 16 + hi * 8);
    }

    // relS[m][r] = Q[q0+m] . rel_k[r]  (r = k - q + WIN, 0..8 valid), via WMMA
    {
        v8f c = zero8();
#pragma unroll
        for (int dc = 0; dc < 2; ++dc) {
            v16bf bf;
#pragma unroll
            for (int j = 0; j < 16; ++j) {
                int d = dc * 32 + hi * 16 + j;
                bf[j] = (ln <= 2 * WIN) ? (bf16)relk[ln * HEAD_CH + d] : (bf16)0.0f;
            }
            c = wmma_bf16(qf[dc], bf, c);
        }
#pragma unroll
        for (int v = 0; v < 8; ++v) relS[v + 8 * hi][ln] = c[v];
    }
    __syncthreads();

    float runmax[8], runsum[8];
    v8f acc[4];
#pragma unroll
    for (int v = 0; v < 8; ++v) { runmax[v] = -1e30f; runsum[v] = 0.0f; }
#pragma unroll
    for (int t = 0; t < 4; ++t) acc[t] = zero8();

    const float scale = 0.125f;  // 1/sqrt(64)

    auto loadK = [&](v16bf (&kfr)[2][2], int kb) {
#pragma unroll
        for (int ks = 0; ks < 2; ++ks)
#pragma unroll
            for (int dc = 0; dc < 2; ++dc)
                kfr[ks][dc] = ld_frag_b(
                    kh + (size_t)(kb + ks * 16 + ln) * HEAD_CH + dc * 32 + hi * 16);
    };

    // One 32-key tile: scores (WMMA) + band rel add + mask + online softmax +
    // P@V from the TDM-filled LDS tile + rel_v band tap.
    auto process_tile = [&](int kb, v16bf (&kfr)[2][2], const bf16* vtile, bool pending) {
        float s0[8], s1[8];
        float mv0 = mask[(size_t)b * SEQ + kb + ln];
        float mv1 = mask[(size_t)b * SEQ + kb + 16 + ln];
#pragma unroll
        for (int ks = 0; ks < 2; ++ks) {
            v8f c = zero8();
#pragma unroll
            for (int dc = 0; dc < 2; ++dc)
                c = wmma_bf16(qf[dc], kfr[ks][dc], c);
            int kcol = kb + ks * 16 + ln;
            float mv = ks ? mv1 : mv0;
#pragma unroll
            for (int v = 0; v < 8; ++v) {
                int qrow = q0 + v + 8 * hi;
                float s = c[v];
                int r = kcol - qrow + WIN;
                if (r >= 0 && r <= 2 * WIN) s += relS[v + 8 * hi][r];
                s *= scale;
                if (mv == 0.0f) s = -1e30f;
                if (ks) s1[v] = s; else s0[v] = s;
            }
        }

#pragma unroll
        for (int v = 0; v < 8; ++v) {
            float t = fmaxf(s0[v], s1[v]);
#pragma unroll
            for (int off = 1; off < 16; off <<= 1)
                t = fmaxf(t, __shfl_xor(t, off, 32));
            float nm = fmaxf(runmax[v], t);
            float corr = __expf(runmax[v] - nm);
            runmax[v] = nm;
            float p0 = __expf(s0[v] - nm);
            float p1 = __expf(s1[v] - nm);
            float rs = p0 + p1;
#pragma unroll
            for (int off = 1; off < 16; off <<= 1)
                rs += __shfl_xor(rs, off, 32);
            runsum[v] = runsum[v] * corr + rs;
#pragma unroll
            for (int dt = 0; dt < 4; ++dt) acc[dt][v] *= corr;
            P_lds[v + 8 * hi][ln]      = (bf16)p0;
            P_lds[v + 8 * hi][16 + ln] = (bf16)p1;
        }
        __syncthreads();

        // wait for this tile's TDM (allow the prefetched one to stay in flight)
        if (pending) __builtin_amdgcn_s_wait_tensorcnt((short)1);
        else         __builtin_amdgcn_s_wait_tensorcnt((short)0);
        asm volatile("" ::: "memory");   // TDM wrote LDS behind compiler's back

        {
            const bf16* pr = &P_lds[ln][0];
            v16bf pf = ld_frag_a(pr + hi * 8, pr + 16 + hi * 8);
#pragma unroll
            for (int dt = 0; dt < 4; ++dt) {
                const bf16* vr = vtile + (size_t)(dt * 16 + ln) * 32 + hi * 16;
                acc[dt] = wmma_bf16(pf, ld_frag_b(vr), acc[dt]);
            }
        }

        if (kb + 31 >= q0 - WIN && kb <= q0 + 15 + WIN) {
#pragma unroll
            for (int v = 0; v < 8; ++v) {
                int qrow = q0 + v + 8 * hi;
#pragma unroll
                for (int r = 0; r <= 2 * WIN; ++r) {
                    int kk = qrow + r - WIN;
                    if (kk >= kb && kk < kb + 32 && kk >= 0 && kk < SEQ) {
                        float w = (float)P_lds[v + 8 * hi][kk - kb];
#pragma unroll
                        for (int dt = 0; dt < 4; ++dt)
                            acc[dt][v] += w * relv[r * HEAD_CH + dt * 16 + ln];
                    }
                }
            }
        }
        __syncthreads();
    };

    // prologue: K frags + TDM of V tile 0 into LDS buffer 0
    v16bf kfA[2][2], kfB[2][2];
    loadK(kfA, 0);
    tdm_load_2d(vt_off0, vh, 32, HEAD_CH, SEQ);

#pragma unroll 1
    for (int kb = 0; kb < SEQ; kb += 64) {
        // tile kb: prefetch tile kb+32 (regs B, LDS buf 1), consume buf 0
        tdm_load_2d(vt_off1, vh + kb + 32, 32, HEAD_CH, SEQ);
        loadK(kfB, kb + 32);
        process_tile(kb, kfA, &Vt[0][0][0], true);

        // tile kb+32: prefetch tile kb+64 (regs A, LDS buf 0), consume buf 1
        bool more = (kb + 64) < SEQ;
        if (more) {
            tdm_load_2d(vt_off0, vh + kb + 64, 32, HEAD_CH, SEQ);
            loadK(kfA, kb + 64);
        }
        process_tile(kb + 32, kfB, &Vt[1][0][0], more);
    }

    // ---- normalize and store context as [b][s][c]
#pragma unroll
    for (int v = 0; v < 8; ++v) {
        float inv = 1.0f / runsum[v];
        int qrow = q0 + v + 8 * hi;
#pragma unroll
        for (int dt = 0; dt < 4; ++dt) {
            ctx[((size_t)b * SEQ + qrow) * IN_CH + h * HEAD_CH + dt * 16 + ln] =
                (bf16)(acc[dt][v] * inv);
        }
    }
}

// ---------------------------------------------------------------- launch
extern "C" void kernel_launch(void* const* d_in, const int* in_sizes, int n_in,
                              void* d_out, int out_size, void* d_ws, size_t ws_size,
                              hipStream_t stream) {
    const float* query = (const float*)d_in[0];
    const float* key   = (const float*)d_in[1];
    const float* value = (const float*)d_in[2];
    const float* mask  = (const float*)d_in[3];
    const float* wq    = (const float*)d_in[4];
    const float* bq    = (const float*)d_in[5];
    const float* wk    = (const float*)d_in[6];
    const float* bk    = (const float*)d_in[7];
    const float* wv    = (const float*)d_in[8];
    const float* bv    = (const float*)d_in[9];
    const float* wo    = (const float*)d_in[10];
    const float* bo    = (const float*)d_in[11];
    const float* rel_k = (const float*)d_in[12];
    const float* rel_v = (const float*)d_in[13];

    char* ws = (char*)d_ws;
    const size_t XSZ = (size_t)BATCH * SEQ * IN_CH * sizeof(bf16);   // 8 MiB each
    const size_t WSZ = (size_t)IN_CH * IN_CH * sizeof(bf16);         // 512 KiB each
    bf16* Xq = (bf16*)(ws + 0 * XSZ);
    bf16* Xk = (bf16*)(ws + 1 * XSZ);
    bf16* Xv = (bf16*)(ws + 2 * XSZ);
    bf16* Wq = (bf16*)(ws + 3 * XSZ + 0 * WSZ);
    bf16* Wk = (bf16*)(ws + 3 * XSZ + 1 * WSZ);
    bf16* Wv = (bf16*)(ws + 3 * XSZ + 2 * WSZ);
    bf16* Wo = (bf16*)(ws + 3 * XSZ + 3 * WSZ);
    bf16* Qb = (bf16*)(ws + 3 * XSZ + 4 * WSZ);
    bf16* Kb = (bf16*)(ws + 4 * XSZ + 4 * WSZ);
    bf16* Vb = (bf16*)(ws + 5 * XSZ + 4 * WSZ);
    bf16* Ct = (bf16*)(ws + 6 * XSZ + 4 * WSZ);

    const int nW = IN_CH * IN_CH;
    cvt_f32_bf16<<<(nW + 255) / 256, 256, 0, stream>>>(wq, Wq, nW);
    cvt_f32_bf16<<<(nW + 255) / 256, 256, 0, stream>>>(wk, Wk, nW);
    cvt_f32_bf16<<<(nW + 255) / 256, 256, 0, stream>>>(wv, Wv, nW);
    cvt_f32_bf16<<<(nW + 255) / 256, 256, 0, stream>>>(wo, Wo, nW);

    dim3 tg(SEQ / 32, IN_CH / 32, BATCH), tb(32, 8);
    transpose_cvt<<<tg, tb, 0, stream>>>(query, Xq);
    transpose_cvt<<<tg, tb, 0, stream>>>(key,   Xk);
    transpose_cvt<<<tg, tb, 0, stream>>>(value, Xv);

    dim3 pg(SEQ / 64, IN_CH / 128, BATCH);
    proj_gemm<<<pg, 128, 0, stream>>>(Wq, Xq, bq, Qb, 0);
    proj_gemm<<<pg, 128, 0, stream>>>(Wk, Xk, bk, Kb, 0);
    proj_gemm<<<pg, 128, 0, stream>>>(Wv, Xv, bv, Vb, 1);

    attn_kernel<<<dim3(SEQ / 16, HEADS, BATCH), 32, 0, stream>>>(
        Qb, Kb, Vb, mask, rel_k, rel_v, Ct);

    out_gemm<<<pg, 128, 0, stream>>>(Wo, Ct, bo, (float*)d_out);
}
